// SimplifiedLlama4Attention_22625887716090
// MI455X (gfx1250) — matrix operand order repaired and checked by reading, not compiled, not run on hardware
//
#include <hip/hip_runtime.h>
#include <hip/hip_bf16.h>
#include <math.h>
#include <stdint.h>

// ---------------------------------------------------------------------------
// Types for CDNA5 WMMA (wave32, 16x16x32 bf16 -> f32)
// ---------------------------------------------------------------------------
typedef __attribute__((ext_vector_type(16))) __bf16 v16bf;
typedef __attribute__((ext_vector_type(8)))  __bf16 v8bf;
typedef __attribute__((ext_vector_type(8)))  float  v8f;

union BF16x16 { v16bf v; v8bf h[2]; __bf16 e[16]; };

constexpr int B_   = 2;
constexpr int S_   = 2048;
constexpr int D_   = 4096;
constexpr int NH_  = 32;
constexpr int NKV_ = 8;
constexpr int HD_  = 128;
constexpr float SCALE_ = 0.08838834764831845f;    // 1/sqrt(128)
constexpr float LOG_THETA_ = 13.122363377404328f; // ln(500000)

// LDS byte offset of a __shared__ pointer (generic ptr low 32 bits = LDS addr)
__device__ __forceinline__ unsigned lds_off(const void* p) {
    return (unsigned)(uintptr_t)p;
}

// CDNA5 async DMA: 16B global -> LDS, tracked by ASYNCcnt (no VGPR round-trip)
__device__ __forceinline__ void async_copy_b128(unsigned lds_addr, const void* gptr) {
    asm volatile("global_load_async_to_lds_b128 %0, %1, off"
                 :: "v"(lds_addr), "v"((unsigned long long)(uintptr_t)gptr)
                 : "memory");
}
__device__ __forceinline__ void wait_async() {
    asm volatile("s_wait_asynccnt 0" ::: "memory");
}

// ---------------------------------------------------------------------------
// fp32 -> bf16 convert
// ---------------------------------------------------------------------------
__global__ void cvt_f32_bf16(const float* __restrict__ src, __bf16* __restrict__ dst, size_t n) {
    size_t i = (size_t)blockIdx.x * blockDim.x + threadIdx.x;
    if (i < n) dst[i] = (__bf16)src[i];
}

// ---------------------------------------------------------------------------
// GEMM: C[M,N] = A[M,K] * W[N,K]^T   (bf16 inputs, f32 accum via WMMA)
// Block: 256 threads = 8 waves (4x2), block tile 128x128, wave tile 32x64,
// K-step 32.  Tiles staged with GLOBAL_LOAD_ASYNC_TO_LDS_B128.
// ---------------------------------------------------------------------------
__global__ void gemm_bf16(const __bf16* __restrict__ A, const __bf16* __restrict__ W,
                          float* __restrict__ C, int M, int N, int K) {
    __shared__ __attribute__((aligned(16))) __bf16 As[128 * 32];
    __shared__ __attribute__((aligned(16))) __bf16 Ws[128 * 32];

    const int tid  = threadIdx.x;   // 0..255
    const int lane = tid & 31;
    const int wid  = tid >> 5;      // 0..7
    const int wy   = wid >> 1;      // 0..3  (M direction)
    const int wx   = wid & 1;       // 0..1  (N direction)
    const int lh   = lane >> 4;     // lane half
    const int ll   = lane & 15;
    const int m0   = blockIdx.y * 128;
    const int n0   = blockIdx.x * 128;

    v8f acc[2][4] = {};

    for (int k0 = 0; k0 < K; k0 += 32) {
        // Stage 128x32 A and 128x32 W tiles: 512 x 16B chunks each,
        // 2 chunks per thread per matrix, direct-to-LDS async DMA.
#pragma unroll
        for (int c = tid; c < 512; c += 256) {
            const int row = c >> 2, q = c & 3;
            async_copy_b128(lds_off(&As[row * 32 + q * 8]),
                            &A[(size_t)(m0 + row) * K + k0 + q * 8]);
            async_copy_b128(lds_off(&Ws[row * 32 + q * 8]),
                            &W[(size_t)(n0 + row) * K + k0 + q * 8]);
        }
        wait_async();
        __syncthreads();

        BF16x16 a[2], b[4];
#pragma unroll
        for (int t = 0; t < 2; ++t) {
            const int r = wy * 32 + t * 16 + ll;       // A row (M)
            a[t].h[0] = *(const v8bf*)&As[r * 32 + 8 * lh];
            a[t].h[1] = *(const v8bf*)&As[r * 32 + 16 + 8 * lh];
        }
#pragma unroll
        for (int u = 0; u < 4; ++u) {
            const int r = wx * 64 + u * 16 + ll;       // W row (N)
            b[u].h[0] = *(const v8bf*)&Ws[r * 32 + 8 * lh];
            b[u].h[1] = *(const v8bf*)&Ws[r * 32 + 16 + 8 * lh];
        }
#pragma unroll
        for (int t = 0; t < 2; ++t)
#pragma unroll
            for (int u = 0; u < 4; ++u)
                acc[t][u] = __builtin_amdgcn_wmma_f32_16x16x32_bf16(
                    false, a[t].v, false, b[u].v, (short)0, acc[t][u], false, false);
        __syncthreads();
    }

#pragma unroll
    for (int t = 0; t < 2; ++t)
#pragma unroll
        for (int u = 0; u < 4; ++u)
#pragma unroll
            for (int r = 0; r < 8; ++r) {
                const int mm = m0 + wy * 32 + t * 16 + r + 8 * lh;
                const int nn = n0 + wx * 64 + u * 16 + ll;
                C[(size_t)mm * N + nn] = acc[t][u][r];
            }
}

// ---------------------------------------------------------------------------
// RoPE (interleaved pairs) + L2 norm over head dim; fp32 [B,S,H,HD] ->
// bf16 [B,H,S,HD].  One block (64 threads) per (b,s,h) row; thread i owns
// pair (2i, 2i+1).
// ---------------------------------------------------------------------------
__global__ void rope_norm(const float* __restrict__ src, __bf16* __restrict__ dst, int H) {
    const int idx = blockIdx.x;            // (b*S + s)*H + h
    const int h   = idx % H;
    const int bs  = idx / H;
    const int s   = bs & (S_ - 1);
    const int b   = bs >> 11;              // S_ = 2048
    const int i   = threadIdx.x;           // 0..63

    const float* row = src + (size_t)idx * HD_;
    const float xe = row[2 * i];
    const float xo = row[2 * i + 1];

    const float inv = __expf(-((float)(2 * i) / (float)HD_) * LOG_THETA_);
    const float fr  = (float)s * inv;
    float sn, cs;
    __sincosf(fr, &sn, &cs);
    const float re = xe * cs - xo * sn;
    const float im = xe * sn + xo * cs;

    __shared__ float red[64];
    red[i] = re * re + im * im;
    __syncthreads();
#pragma unroll
    for (int st = 32; st > 0; st >>= 1) {
        if (i < st) red[i] += red[i + st];
        __syncthreads();
    }
    const float scale = rsqrtf(red[0] * (1.0f / (float)HD_) + 1e-6f);

    __bf16* drow = dst + (((size_t)(b * H + h)) * S_ + s) * HD_;
    drow[2 * i]     = (__bf16)(re * scale);
    drow[2 * i + 1] = (__bf16)(im * scale);
}

// ---------------------------------------------------------------------------
// V fp32 [B,S,KVH,HD] -> bf16 transposed [B,KVH,HD,S]
// ---------------------------------------------------------------------------
__global__ void v_transpose(const float* __restrict__ Vf, __bf16* __restrict__ VbfT) {
    const size_t i  = (size_t)blockIdx.x * blockDim.x + threadIdx.x;  // B*S*KVH*HD
    const int    hd = (int)(i & (HD_ - 1));
    const size_t t  = i >> 7;
    const int    h  = (int)(t & (NKV_ - 1));
    const size_t bs = t >> 3;
    const int    s  = (int)(bs & (S_ - 1));
    const int    b  = (int)(bs >> 11);
    VbfT[(((size_t)(b * NKV_ + h)) * HD_ + hd) * S_ + s] = (__bf16)Vf[i];
}

// ---------------------------------------------------------------------------
// Attention pass 1: per-wave 16-row q tile, online softmax stats (max, sum).
// Block = 256 threads = 8 independent waves.
// ---------------------------------------------------------------------------
__global__ void attn_pass1(const __bf16* __restrict__ Qbf, const __bf16* __restrict__ Kbf,
                           float* __restrict__ rowm, float* __restrict__ rowsum) {
    const int lane = threadIdx.x & 31;
    const int wid  = threadIdx.x >> 5;
    const int gw   = blockIdx.x * 8 + wid;       // 0..8191
    const int qt   = gw & 127;
    const int h    = (gw >> 7) & 31;
    const int b    = gw >> 12;
    const int kvh  = h >> 2;
    const int q0   = qt * 16;
    const int lh   = lane >> 4, ll = lane & 15;

    const __bf16* qrow = Qbf + (((size_t)(b * NH_ + h)) * S_ + (q0 + ll)) * HD_;
    BF16x16 qf[4];
#pragma unroll
    for (int f = 0; f < 4; ++f) {
        qf[f].h[0] = *(const v8bf*)(qrow + 32 * f + 8 * lh);
        qf[f].h[1] = *(const v8bf*)(qrow + 32 * f + 16 + 8 * lh);
    }

    float m[8], s[8];
#pragma unroll
    for (int r = 0; r < 8; ++r) { m[r] = -3e38f; s[r] = 0.f; }

    const __bf16* kbase = Kbf + ((size_t)(b * NKV_ + kvh)) * S_ * HD_;
    for (int kc = 0; kc <= qt; ++kc) {
        const int k0 = kc * 16;
        v8f sc = {};
        const __bf16* krow = kbase + (size_t)(k0 + ll) * HD_ + 8 * lh;
#pragma unroll
        for (int f = 0; f < 4; ++f) {
            BF16x16 kf;
            kf.h[0] = *(const v8bf*)(krow + 32 * f);
            kf.h[1] = *(const v8bf*)(krow + 32 * f + 16);
            sc = __builtin_amdgcn_wmma_f32_16x16x32_bf16(
                false, qf[f].v, false, kf.v, (short)0, sc, false, false);
        }
        const int col = k0 + ll;
#pragma unroll
        for (int r = 0; r < 8; ++r) {
            const int row = q0 + r + 8 * lh;
            const float v  = (col <= row) ? sc[r] * SCALE_ : -3e38f;
            const float nm = fmaxf(m[r], v);
            const float p  = (v    < -1e37f) ? 0.f : __expf(v - nm);
            const float al = (m[r] < -1e37f) ? 0.f : __expf(m[r] - nm);
            s[r] = s[r] * al + p;
            m[r] = nm;
        }
    }

    // Merge across the 16 lanes sharing each row-half.
#pragma unroll
    for (int mask = 1; mask < 16; mask <<= 1) {
#pragma unroll
        for (int r = 0; r < 8; ++r) {
            const float om = __shfl_xor(m[r], mask, 32);
            const float os = __shfl_xor(s[r], mask, 32);
            const float nm = fmaxf(m[r], om);
            const float a0 = (m[r] < -1e37f) ? 0.f : __expf(m[r] - nm);
            const float a1 = (om   < -1e37f) ? 0.f : __expf(om - nm);
            s[r] = s[r] * a0 + os * a1;
            m[r] = nm;
        }
    }
    if (ll == 0) {
#pragma unroll
        for (int r = 0; r < 8; ++r) {
            const size_t o = (size_t)(b * NH_ + h) * S_ + q0 + r + 8 * lh;
            rowm[o]   = m[r];
            rowsum[o] = s[r];
        }
    }
}

// ---------------------------------------------------------------------------
// Attention pass 2: recompute scores, emit normalized probs (fp32) to attn
// output, re-layout P via LDS into WMMA A-fragments, accumulate O = P*V.
// ---------------------------------------------------------------------------
__global__ void attn_pass2(const __bf16* __restrict__ Qbf, const __bf16* __restrict__ Kbf,
                           const __bf16* __restrict__ VbfT,
                           const float* __restrict__ rowm, const float* __restrict__ rowsum,
                           float* __restrict__ attn, __bf16* __restrict__ Obf) {
    __shared__ float Pst[8][16 * 32];   // per-wave P staging (2KB each)

    const int lane = threadIdx.x & 31;
    const int wid  = threadIdx.x >> 5;
    const int gw   = blockIdx.x * 8 + wid;
    const int qt   = gw & 127;
    const int h    = (gw >> 7) & 31;
    const int b    = gw >> 12;
    const int kvh  = h >> 2;
    const int q0   = qt * 16;
    const int lh   = lane >> 4, ll = lane & 15;

    const __bf16* qrow = Qbf + (((size_t)(b * NH_ + h)) * S_ + (q0 + ll)) * HD_;
    BF16x16 qf[4];
#pragma unroll
    for (int f = 0; f < 4; ++f) {
        qf[f].h[0] = *(const v8bf*)(qrow + 32 * f + 8 * lh);
        qf[f].h[1] = *(const v8bf*)(qrow + 32 * f + 16 + 8 * lh);
    }

    float rm[8], rs[8];
#pragma unroll
    for (int r = 0; r < 8; ++r) {
        const size_t o = (size_t)(b * NH_ + h) * S_ + q0 + r + 8 * lh;
        rm[r] = rowm[o];
        rs[r] = 1.0f / rowsum[o];
    }

    v8f ov[8] = {};
    const __bf16* kbase = Kbf + ((size_t)(b * NKV_ + kvh)) * S_ * HD_;
    const __bf16* vbase = VbfT + ((size_t)(b * NKV_ + kvh)) * (size_t)HD_ * S_;
    const size_t  arowstride = (size_t)S_;
    float* abase = attn + ((size_t)(b * NH_ + h)) * S_ * S_;

    for (int kc = 0; kc < 64; ++kc) {
        const int k0 = kc * 32;
        if (k0 > q0 + 15) {
            // fully masked: write zeros to attn
#pragma unroll
            for (int r = 0; r < 8; ++r) {
                float* ap = abase + (size_t)(q0 + r + 8 * lh) * arowstride + k0;
                ap[ll] = 0.f;
                ap[16 + ll] = 0.f;
            }
            continue;
        }
        // scores for cols [k0, k0+16) and [k0+16, k0+32)
        v8f sc0 = {}, sc1 = {};
        const __bf16* krow0 = kbase + (size_t)(k0 + ll) * HD_ + 8 * lh;
        const __bf16* krow1 = kbase + (size_t)(k0 + 16 + ll) * HD_ + 8 * lh;
#pragma unroll
        for (int f = 0; f < 4; ++f) {
            BF16x16 kf0, kf1;
            kf0.h[0] = *(const v8bf*)(krow0 + 32 * f);
            kf0.h[1] = *(const v8bf*)(krow0 + 32 * f + 16);
            kf1.h[0] = *(const v8bf*)(krow1 + 32 * f);
            kf1.h[1] = *(const v8bf*)(krow1 + 32 * f + 16);
            sc0 = __builtin_amdgcn_wmma_f32_16x16x32_bf16(
                false, qf[f].v, false, kf0.v, (short)0, sc0, false, false);
            sc1 = __builtin_amdgcn_wmma_f32_16x16x32_bf16(
                false, qf[f].v, false, kf1.v, (short)0, sc1, false, false);
        }
        // normalized probabilities -> attn output + LDS staging
#pragma unroll
        for (int r = 0; r < 8; ++r) {
            const int row = q0 + r + 8 * lh;
            const float p0 = (k0 + ll <= row)      ? __expf(sc0[r] * SCALE_ - rm[r]) * rs[r] : 0.f;
            const float p1 = (k0 + 16 + ll <= row) ? __expf(sc1[r] * SCALE_ - rm[r]) * rs[r] : 0.f;
            float* ap = abase + (size_t)row * arowstride + k0;
            ap[ll] = p0;
            ap[16 + ll] = p1;
            Pst[wid][(r + 8 * lh) * 32 + ll]      = p0;
            Pst[wid][(r + 8 * lh) * 32 + 16 + ll] = p1;
        }
        asm volatile("s_wait_dscnt 0" ::: "memory");

        // Re-layout P into WMMA A fragment (16x32, bf16)
        BF16x16 pf;
        const float* pb = &Pst[wid][ll * 32];
#pragma unroll
        for (int j = 0; j < 8; ++j) pf.e[j] = (__bf16)pb[8 * lh + j];
#pragma unroll
        for (int j = 0; j < 8; ++j) pf.e[8 + j] = (__bf16)pb[16 + 8 * lh + j];

        // O += P * V  over 8 HD tiles
#pragma unroll
        for (int u = 0; u < 8; ++u) {
            BF16x16 vf;
            const __bf16* vp = vbase + (size_t)(16 * u + ll) * S_ + k0 + 8 * lh;
            vf.h[0] = *(const v8bf*)vp;
            vf.h[1] = *(const v8bf*)(vp + 16);
            ov[u] = __builtin_amdgcn_wmma_f32_16x16x32_bf16(
                false, pf.v, false, vf.v, (short)0, ov[u], false, false);
        }
    }

    // Store O tile: Obf[b][row][h*128 + hd], bf16
#pragma unroll
    for (int u = 0; u < 8; ++u)
#pragma unroll
        for (int r = 0; r < 8; ++r) {
            const int row = q0 + r + 8 * lh;
            Obf[((size_t)b * S_ + row) * (NH_ * HD_) + h * HD_ + 16 * u + ll] = (__bf16)ov[u][r];
        }
}

// ---------------------------------------------------------------------------
// Host-side orchestration
// ---------------------------------------------------------------------------
extern "C" void kernel_launch(void* const* d_in, const int* in_sizes, int n_in,
                              void* d_out, int out_size, void* d_ws, size_t ws_size,
                              hipStream_t stream) {
    (void)in_sizes; (void)n_in; (void)out_size; (void)ws_size;

    const float* X  = (const float*)d_in[0];
    // d_in[1] = attention_mask (pure causal; reproduced analytically)
    // d_in[2] = position_ids   (arange; reproduced analytically)
    const float* Wq = (const float*)d_in[3];
    const float* Wk = (const float*)d_in[4];
    const float* Wv = (const float*)d_in[5];
    const float* Wo = (const float*)d_in[6];

    float* outF = (float*)d_out;                     // [B,S,D] fp32
    float* attn = outF + (size_t)B_ * S_ * D_;       // [B,NH,S,S] fp32

    char* ws = (char*)d_ws;
    // Workspace layout (byte offsets), with dead-buffer aliasing:
    __bf16* Xbf  = (__bf16*)(ws + 0x0000000);   // 32MB ; later Obf
    __bf16* Wqbf = (__bf16*)(ws + 0x2000000);   // 32MB ; later Qbf
    __bf16* Wkbf = (__bf16*)(ws + 0x4000000);   //  8MB ; later Kbf
    __bf16* Wvbf = (__bf16*)(ws + 0x4800000);   //  8MB ; later VbfT
    __bf16* Wobf = (__bf16*)(ws + 0x5000000);   // 32MB
    float*  Qf   = (float*) (ws + 0x7000000);   // 64MB
    float*  Kf   = (float*) (ws + 0xB000000);   // 16MB
    float*  Vf   = (float*) (ws + 0xC000000);   // 16MB
    float*  rowm = (float*) (ws + 0xD000000);   // 512KB
    float*  rsum = (float*) (ws + 0xD080000);   // 512KB
    __bf16* Obf  = Xbf;
    __bf16* Qbf  = Wqbf;
    __bf16* Kbf  = Wkbf;
    __bf16* VbfT = Wvbf;

    const size_t nX  = (size_t)B_ * S_ * D_;          // 16,777,216
    const size_t nWq = (size_t)NH_ * HD_ * D_;        // 16,777,216
    const size_t nWk = (size_t)NKV_ * HD_ * D_;       //  4,194,304
    const size_t nWo = (size_t)D_ * NH_ * HD_;        // 16,777,216

    // 1) fp32 -> bf16 conversions
    cvt_f32_bf16<<<(int)(nX  / 256), 256, 0, stream>>>(X,  Xbf,  nX);
    cvt_f32_bf16<<<(int)(nWq / 256), 256, 0, stream>>>(Wq, Wqbf, nWq);
    cvt_f32_bf16<<<(int)(nWk / 256), 256, 0, stream>>>(Wk, Wkbf, nWk);
    cvt_f32_bf16<<<(int)(nWk / 256), 256, 0, stream>>>(Wv, Wvbf, nWk);
    cvt_f32_bf16<<<(int)(nWo / 256), 256, 0, stream>>>(Wo, Wobf, nWo);

    // 2) QKV projections: C = X * W^T  (M = B*S = 4096, K = 4096)
    gemm_bf16<<<dim3(32, 32), 256, 0, stream>>>(Xbf, Wqbf, Qf, B_ * S_, NH_ * HD_,  D_);
    gemm_bf16<<<dim3( 8, 32), 256, 0, stream>>>(Xbf, Wkbf, Kf, B_ * S_, NKV_ * HD_, D_);
    gemm_bf16<<<dim3( 8, 32), 256, 0, stream>>>(Xbf, Wvbf, Vf, B_ * S_, NKV_ * HD_, D_);

    // 3) RoPE + L2 norm (Q, K); V transpose/convert
    rope_norm<<<B_ * S_ * NH_,  64, 0, stream>>>(Qf, Qbf, NH_);
    rope_norm<<<B_ * S_ * NKV_, 64, 0, stream>>>(Kf, Kbf, NKV_);
    v_transpose<<<(int)((size_t)B_ * S_ * NKV_ * HD_ / 256), 256, 0, stream>>>(Vf, VbfT);

    // 4) softmax statistics (causal, online)
    attn_pass1<<<(B_ * NH_ * (S_ / 16)) / 8, 256, 0, stream>>>(Qbf, Kbf, rowm, rsum);

    // 5) probabilities (-> attn output) + O = P*V
    attn_pass2<<<(B_ * NH_ * (S_ / 16)) / 8, 256, 0, stream>>>(Qbf, Kbf, VbfT, rowm, rsum, attn, Obf);

    // 6) final projection: final = O * Wo^T
    gemm_bf16<<<dim3(32, 32), 256, 0, stream>>>(Obf, Wobf, outF, B_ * S_, D_, NH_ * HD_);
}
